// RNNModel_47373489275228
// MI455X (gfx1250) — matrix operand order repaired and checked by reading, not compiled, not run on hardware
//
#include <hip/hip_runtime.h>
#include <hip/hip_bf16.h>
#include <math.h>

// ---------------------------------------------------------------------------
// Problem constants (from reference)
// ---------------------------------------------------------------------------
#define NTOKEN 32000
#define NINP   1024
#define NHID   1024
#define SEQ    64
#define BSZ    64
#define NSAMP  32
#define NPOS   (SEQ * BSZ)        // 4096
#define BETA   10.0f
#define REGB   1e-4f

typedef unsigned short u16;
typedef __attribute__((ext_vector_type(8)))  u16     u16x8;
typedef __attribute__((ext_vector_type(16))) __bf16  v16bf;
typedef __attribute__((ext_vector_type(8)))  float   v8f;
typedef __attribute__((ext_vector_type(4)))  int     v4i;

struct U16Pair { u16x8 lo; u16x8 hi; };

#define GLOBAL_AS __attribute__((address_space(1)))
#define LDS_AS    __attribute__((address_space(3)))

#if __has_builtin(__builtin_amdgcn_global_load_async_to_lds_b128) && \
    __has_builtin(__builtin_amdgcn_s_wait_asynccnt)
#define USE_ASYNC_STAGE 1
#else
#define USE_ASYNC_STAGE 0
#endif

// ---------------------------------------------------------------------------
// Helpers
// ---------------------------------------------------------------------------
__device__ __forceinline__ u16 f2bf(float f) {
    unsigned u = __builtin_bit_cast(unsigned, f);
    unsigned r = u + 0x7FFFu + ((u >> 16) & 1u);   // round-to-nearest-even
    return (u16)(r >> 16);
}

__device__ __forceinline__ v16bf make_frag(u16x8 lo, u16x8 hi) {
    U16Pair t; t.lo = lo; t.hi = hi;
    return __builtin_bit_cast(v16bf, t);
}

// A-fragment (16x32, M x K), row-major source, row picked per-lane by caller.
// Layout (ISA 7.12.2): lane<16 holds K = {k0+0..7, k0+16..23}; lane>=16 adds 8.
__device__ __forceinline__ v16bf load_a_frag(const u16* rowp, int k0, int lane) {
    const u16* p = rowp + k0 + ((lane >> 4) << 3);
    u16x8 lo = *(const u16x8*)(p);
    u16x8 hi = *(const u16x8*)(p + 16);
    return make_frag(lo, hi);
}

// B-fragment (32x16, K x N), source stored N-major with K contiguous
// (i.e. W[n][k] row-major). lane<16 holds K = {k0+0..7, k0+16..23}; lane>=16 adds 8.
__device__ __forceinline__ v16bf load_b_frag(const u16* colp, int k0, int lane) {
    const u16* p = colp + k0 + ((lane >> 4) << 3);
    u16x8 lo = *(const u16x8*)(p);
    u16x8 hi = *(const u16x8*)(p + 16);
    return make_frag(lo, hi);
}

__device__ __forceinline__ v8f wmma_bf16(v16bf a, v16bf b, v8f c) {
    return __builtin_amdgcn_wmma_f32_16x16x32_bf16(
        /*neg_a=*/false, a, /*neg_b=*/false, b,
        /*c_mod=*/(short)0, c, /*reuse_a=*/false, /*reuse_b=*/false);
}

// ---------------------------------------------------------------------------
// Kernels
// ---------------------------------------------------------------------------

// f32 -> bf16(as u16) conversion
__global__ void k_f2bf(const float* __restrict__ src, u16* __restrict__ dst, int n) {
    int i = blockIdx.x * 256 + threadIdx.x;
    if (i < n) dst[i] = f2bf(src[i]);
}

// init: raw[0] = hidden, Hb[0] = bf16(hidden), d_out = 0, barrier = 0
__global__ void k_init(const float* __restrict__ hidden,
                       float* __restrict__ raw, u16* __restrict__ Hb,
                       float* __restrict__ out, int* __restrict__ bar) {
    int i = blockIdx.x * 256 + threadIdx.x;
    if (i < BSZ * NHID) {
        float h = hidden[i];
        raw[i] = h;
        Hb[i]  = f2bf(h);
    }
    if (i == 0) { out[0] = 0.0f; bar[0] = 0; }
}

// ---------------------------------------------------------------------------
// Persistent RNN scan: one kernel, 32 co-resident blocks, monotonic grid
// barrier between dependent timesteps (launch-latency elimination).
// 256 waves total, one 16x16 tile each per step. The asm memory barrier at
// the top of each step stops LICM from caching 512 VGPRs of W-fragments
// across timesteps (which caused scratch spills); W is L2/L0-resident so
// re-streaming it each step is nearly free.
// ---------------------------------------------------------------------------
__global__ void __launch_bounds__(256)
k_rnn_scan(const int* __restrict__ data,
           const u16* __restrict__ embb, u16* __restrict__ Hb,
           float* __restrict__ raw,
           const u16* __restrict__ Wihb, const u16* __restrict__ Whhb,
           const float* __restrict__ b_ih, const float* __restrict__ b_hh,
           int* __restrict__ bar) {
    int wid  = threadIdx.x >> 5;
    int lane = threadIdx.x & 31;
    int wg   = blockIdx.x * 8 + wid;         // 0..255
    int mtile = wg >> 6;                     // 0..3   (BSZ=64 -> 4 tiles)
    int ntile = wg & 63;                     // 0..63  (NHID=1024 -> 64 tiles)

    int mrow = mtile * 16 + (lane & 15);
    int ncol = ntile * 16 + (lane & 15);

    const u16* bi = Wihb + (size_t)ncol * NINP;
    const u16* bh = Whhb + (size_t)ncol * NHID;
    float bb = b_ih[ncol] + b_hh[ncol];

#pragma clang loop unroll(disable)
    for (int s = 0; s < SEQ; ++s) {
        // opaque memory barrier: forbid hoisting any loads across timesteps
        asm volatile("" ::: "memory");

        int token = data[s * BSZ + mrow];
        const u16* ax = embb + (size_t)token * NINP;
        const u16* ah = Hb   + ((size_t)s * BSZ + mrow) * NHID;

        v8f acc = {};
        for (int kt = 0; kt < NINP / 32; ++kt) {
            int k0 = kt * 32;
            acc = wmma_bf16(load_a_frag(ax, k0, lane), load_b_frag(bi, k0, lane), acc);
            acc = wmma_bf16(load_a_frag(ah, k0, lane), load_b_frag(bh, k0, lane), acc);
        }

        int rbase = (s + 1) * BSZ + mtile * 16 + ((lane >> 4) << 3);
#pragma unroll
        for (int v = 0; v < 8; ++v) {
            float h = tanhf(acc[v] + bb);
            size_t idx = (size_t)(rbase + v) * NHID + ncol;
            raw[idx] = h;
            Hb[idx]  = f2bf(h);
        }

        // ---- grid barrier (monotonic counter, release/acquire at agent scope)
        __threadfence();
        __syncthreads();
        if (threadIdx.x == 0) {
            __hip_atomic_fetch_add(bar, 1, __ATOMIC_RELEASE, __HIP_MEMORY_SCOPE_AGENT);
            while (__hip_atomic_load(bar, __ATOMIC_ACQUIRE, __HIP_MEMORY_SCOPE_AGENT)
                   < 32 * (s + 1)) {
                __builtin_amdgcn_s_sleep(1);
            }
        }
        __syncthreads();
    }
}

// d_pos[r] = BETA * ||raw[r+64] - raw[r]||^2 + bias[data[r]]   (wave per r)
__global__ void k_dpos(const float* __restrict__ raw, const int* __restrict__ data,
                       const float* __restrict__ bias, float* __restrict__ d_pos) {
    int wid  = threadIdx.x >> 5;
    int lane = threadIdx.x & 31;
    int r = blockIdx.x * 8 + wid;            // grid 512 -> 4096 waves
    const float* a = raw + (size_t)(r + BSZ) * NHID;
    const float* b = raw + (size_t)r * NHID;
    float s = 0.0f;
    for (int i = lane; i < NHID; i += 32) {
        float d = a[i] - b[i];
        s += d * d;
    }
    for (int k = 1; k < 32; k <<= 1) s += __shfl_xor(s, k, 32);
    if (lane == 0) d_pos[r] = BETA * s + bias[data[r]];
}

// hU = h_prev @ W_hh^T + b_hh  (4096 x 1024, f32 out).
// One wave computes a 16x64 slab: 4 accumulators share each A-fragment.
__global__ void __launch_bounds__(256)
k_hU(const u16* __restrict__ Hb, const u16* __restrict__ Whhb,
     const float* __restrict__ b_hh, float* __restrict__ hU) {
    int wid  = threadIdx.x >> 5;
    int lane = threadIdx.x & 31;
    int wg   = blockIdx.x * 8 + wid;         // grid 512 -> 4096 waves
    int mtile  = wg >> 4;                    // 0..255
    int ngroup = wg & 15;                    // 0..15 (4 tiles = 64 cols each)

    int mrow = mtile * 16 + (lane & 15);
    const u16* a_row = Hb + (size_t)mrow * NHID;   // rows 0..4095 == h_prev

    const u16* bcol[4];
    int ncol[4];
#pragma unroll
    for (int c = 0; c < 4; ++c) {
        ncol[c] = (ngroup * 4 + c) * 16 + (lane & 15);
        bcol[c] = Whhb + (size_t)ncol[c] * NHID;
    }

    v8f acc[4] = {};
    for (int kt = 0; kt < NHID / 32; ++kt) {
        int k0 = kt * 32;
        v16bf a = load_a_frag(a_row, k0, lane);
#pragma unroll
        for (int c = 0; c < 4; ++c)
            acc[c] = wmma_bf16(a, load_b_frag(bcol[c], k0, lane), acc[c]);
    }

    int rbase = mtile * 16 + ((lane >> 4) << 3);
#pragma unroll
    for (int c = 0; c < 4; ++c) {
        float bb = b_hh[ncol[c]];
#pragma unroll
        for (int v = 0; v < 8; ++v)
            hU[(size_t)(rbase + v) * NHID + ncol[c]] = acc[c][v] + bb;
    }
}

// Fused negatives: block (rblock, n) stages a 16-row bf16 A-panel in LDS
// (async-to-LDS), then 8 waves each compute two 16x64 slabs (4 accumulators
// per A-fragment), fusing +hU +b_ih, tanh, squared-diff against h_prev,
// cross-lane reduce, LDS atomics -> d_neg.
__global__ void __launch_bounds__(256)
k_neg(const int* __restrict__ samples, const u16* __restrict__ embb,
      const u16* __restrict__ Wihb, const float* __restrict__ b_ih,
      const float* __restrict__ hU, const float* __restrict__ raw,
      const float* __restrict__ bias, float* __restrict__ d_neg) {
    __shared__ u16  As[16 * NINP];           // 32 KB A-panel
    __shared__ float dsum[16];

    int n = blockIdx.y;
    int rbase = blockIdx.x * 16;
    int t = threadIdx.x;

    // cooperative stage: 16 threads per row, 64 u16 each (8 x b128)
    {
        int row = t >> 4, chunk = t & 15;
        int tok = samples[n * NPOS + rbase + row];
        const u16* src = embb + (size_t)tok * NINP + chunk * 64;
        u16*       dst = As + row * NINP + chunk * 64;
#if USE_ASYNC_STAGE
#pragma unroll
        for (int i = 0; i < 8; ++i)
            __builtin_amdgcn_global_load_async_to_lds_b128(
                (GLOBAL_AS v4i*)(src + i * 8),
                (LDS_AS v4i*)(dst + i * 8), 0, 0);
        __builtin_amdgcn_s_wait_asynccnt(0);
#else
#pragma unroll
        for (int i = 0; i < 8; ++i)
            *(u16x8*)(dst + i * 8) = *(const u16x8*)(src + i * 8);
#endif
    }
    if (t < 16) dsum[t] = 0.0f;
    __syncthreads();

    int wid = t >> 5, lane = t & 31;
    const u16* arow = As + (lane & 15) * NINP;
    int rowoff = ((lane >> 4) << 3);

    float part[8];
#pragma unroll
    for (int v = 0; v < 8; ++v) part[v] = 0.0f;

    for (int g = 0; g < 2; ++g) {                 // 2 slabs of 4 N-tiles
        const u16* bcol[4];
        int ncol[4];
#pragma unroll
        for (int c = 0; c < 4; ++c) {
            ncol[c] = (wid * 8 + g * 4 + c) * 16 + (lane & 15);
            bcol[c] = Wihb + (size_t)ncol[c] * NINP;
        }
        v8f acc[4] = {};
        for (int kt = 0; kt < NINP / 32; ++kt) {
            int k0 = kt * 32;
            v16bf a = load_a_frag(arow, k0, lane);
#pragma unroll
            for (int c = 0; c < 4; ++c)
                acc[c] = wmma_bf16(a, load_b_frag(bcol[c], k0, lane), acc[c]);
        }
#pragma unroll
        for (int c = 0; c < 4; ++c) {
            float bi = b_ih[ncol[c]];
#pragma unroll
            for (int v = 0; v < 8; ++v) {
                int r = rbase + rowoff + v;
                size_t idx = (size_t)r * NHID + ncol[c];
                float o = tanhf(acc[c][v] + bi + hU[idx]);
                float d = raw[idx] - o;           // h_prev - out
                part[v] += d * d;
            }
        }
    }
    // reduce over the 16 lane-columns within each half-wave
    for (int k = 1; k < 16; k <<= 1) {
#pragma unroll
        for (int v = 0; v < 8; ++v) part[v] += __shfl_xor(part[v], k, 32);
    }
    if ((lane & 15) == 0) {
#pragma unroll
        for (int v = 0; v < 8; ++v) atomicAdd(&dsum[rowoff + v], part[v]);
    }
    __syncthreads();
    if (t < 16) {
        int r = rbase + t;
        int tok = samples[n * NPOS + r];
        d_neg[n * NPOS + r] = BETA * dsum[t] + bias[tok];
    }
}

// per-position NLL with 33-way log-softmax; mean over (S,B) weighted by binary
__global__ void k_loss(const float* __restrict__ d_pos, const float* __restrict__ d_neg,
                       const float* __restrict__ binary, float* __restrict__ out) {
    int r = blockIdx.x * 256 + threadIdx.x;      // grid 16 -> 4096 threads
    float c = 0.0f;
    if (r < NPOS) {
        float x0 = -d_pos[r];
        float xs[NSAMP];
        float m = x0;
#pragma unroll
        for (int j = 0; j < NSAMP; ++j) {
            xs[j] = -d_neg[j * NPOS + r];
            m = fmaxf(m, xs[j]);
        }
        float se = expf(x0 - m);
#pragma unroll
        for (int j = 0; j < NSAMP; ++j) se += expf(xs[j] - m);
        float lse = m + logf(se);
        c = (lse - x0) * binary[r] * (1.0f / (float)NPOS);
    }
    for (int k = 1; k < 32; k <<= 1) c += __shfl_xor(c, k, 32);
    if ((threadIdx.x & 31) == 0) atomicAdd(out, c);
}

// + REG_BIAS * sum(bias^2)
__global__ void k_breg(const float* __restrict__ bias, float* __restrict__ out) {
    int i = blockIdx.x * 256 + threadIdx.x;
    float c = 0.0f;
    if (i < NTOKEN) {
        float v = bias[i];
        c = REGB * v * v;
    }
    for (int k = 1; k < 32; k <<= 1) c += __shfl_xor(c, k, 32);
    if ((threadIdx.x & 31) == 0) atomicAdd(out, c);
}

// ---------------------------------------------------------------------------
// Host launcher
// ---------------------------------------------------------------------------
extern "C" void kernel_launch(void* const* d_in, const int* in_sizes, int n_in,
                              void* d_out, int out_size, void* d_ws, size_t ws_size,
                              hipStream_t stream) {
    (void)in_sizes; (void)n_in; (void)out_size; (void)ws_size;

    const int*   data    = (const int*)  d_in[0];
    const float* binary  = (const float*)d_in[1];
    const float* hidden  = (const float*)d_in[2];
    const int*   samples = (const int*)  d_in[3];
    const float* emb     = (const float*)d_in[4];
    const float* W_ih    = (const float*)d_in[5];
    const float* b_ih    = (const float*)d_in[6];
    const float* W_hh    = (const float*)d_in[7];
    const float* b_hh    = (const float*)d_in[8];
    const float* bias    = (const float*)d_in[9];
    float* out = (float*)d_out;

    // workspace carve (all region sizes are multiples of 256B)
    char* p = (char*)d_ws;
    u16*   embb  = (u16*)p;   p += (size_t)NTOKEN * NINP * sizeof(u16);      // 64 MB
    u16*   Wihb  = (u16*)p;   p += (size_t)NHID * NINP * sizeof(u16);        // 2 MB
    u16*   Whhb  = (u16*)p;   p += (size_t)NHID * NHID * sizeof(u16);        // 2 MB
    float* raw   = (float*)p; p += (size_t)(SEQ + 1) * BSZ * NHID * sizeof(float); // 17 MB
    u16*   Hb    = (u16*)p;   p += (size_t)(SEQ + 1) * BSZ * NHID * sizeof(u16);   // 8.5 MB
    float* hU    = (float*)p; p += (size_t)NPOS * NHID * sizeof(float);      // 16.8 MB
    float* d_pos = (float*)p; p += (size_t)NPOS * sizeof(float);
    float* d_neg = (float*)p; p += (size_t)NSAMP * NPOS * sizeof(float);
    int*   bar   = (int*)p;   p += 256;

    // 1) bf16 conversions (L2-resident afterwards)
    {
        int n = NTOKEN * NINP;
        k_f2bf<<<(n + 255) / 256, 256, 0, stream>>>(emb, embb, n);
    }
    {
        int n = NHID * NINP;
        k_f2bf<<<(n + 255) / 256, 256, 0, stream>>>(W_ih, Wihb, n);
        k_f2bf<<<(n + 255) / 256, 256, 0, stream>>>(W_hh, Whhb, n);
    }

    // 2) init h0 + zero output accumulator + grid-barrier counter
    k_init<<<(BSZ * NHID + 255) / 256, 256, 0, stream>>>(hidden, raw, Hb, out, bar);

    // 3) persistent sequential RNN scan (64 dependent steps, 1 launch)
    k_rnn_scan<<<32, 256, 0, stream>>>(data, embb, Hb, raw, Wihb, Whhb, b_ih, b_hh, bar);

    // 4) positive distances
    k_dpos<<<NPOS / 8, 256, 0, stream>>>(raw, data, bias, d_pos);

    // 5) hU = h_prev @ W_hh^T + b_hh
    k_hU<<<512, 256, 0, stream>>>(Hb, Whhb, b_hh, hU);

    // 6) fused negatives (dominant GEMM + tanh + distance reduction)
    k_neg<<<dim3(NPOS / 16, NSAMP), 256, 0, stream>>>(samples, embb, Wihb, b_ih,
                                                      hU, raw, bias, d_neg);

    // 7) softmax / NLL loss + bias regularizer
    k_loss<<<NPOS / 256, 256, 0, stream>>>(d_pos, d_neg, binary, out);
    k_breg<<<(NTOKEN + 255) / 256, 256, 0, stream>>>(bias, out);
}
